// SelfAttention_16389595201615
// MI455X (gfx1250) — matrix operand order repaired and checked by reading, not compiled, not run on hardware
//
#include <hip/hip_runtime.h>
#include <hip/hip_bf16.h>

// ---------------------------------------------------------------------------
// Self-attention (q=k=v = x@Wq+bq), B=8, S=2048, D=1024, all via bf16 WMMA.
// Pipeline: WqT(bf16) -> proj(q_bf + qT_bf) -> scores(q@q^T, bf16)
//           -> softmax(rows, in place) -> z = attn @ q (f32 out).
// All WMMA operand loads are contiguous 16-byte vector loads.
// ---------------------------------------------------------------------------

typedef __bf16 bf16_t;
typedef __attribute__((ext_vector_type(16))) __bf16 v16bf;
typedef __attribute__((ext_vector_type(8)))  float  v8f;
typedef __attribute__((ext_vector_type(4)))  float  v4f;

#define ATT_B 8
#define ATT_S 2048
#define ATT_D 1024

__device__ __forceinline__ v8f wmma_bf16(v16bf a, v16bf b, v8f c) {
    return __builtin_amdgcn_wmma_f32_16x16x32_bf16(
        /*neg_a=*/false, a, /*neg_b=*/false, b,
        /*c_mod=*/(short)0, c, /*reuse_a=*/false, /*reuse_b=*/false);
}

// A fragment 16x32 from row-major bf16 [rows, ld].
// Lane L: row = row0 + (L&15); VGPR0-3 hold K = k0 + 8*(L>>4) + 0..7,
// VGPR4-7 hold K = k0 + 16 + 8*(L>>4) + 0..7  -> two 16-byte loads.
__device__ __forceinline__ v16bf load_frag_a_bf16(const bf16_t* p, int row0,
                                                  int k0, int ld, int lane) {
    const int m = lane & 15;
    const int h = lane >> 4;
    const bf16_t* base = p + (size_t)(row0 + m) * ld + k0 + 8 * h;
    union { v16bf v; uint4 u[2]; } f;
    f.u[0] = *(const uint4*)(base);
    f.u[1] = *(const uint4*)(base + 16);
    return f.v;
}

// A fragment 16x32 from row-major f32 (convert to bf16 in regs).
__device__ __forceinline__ v16bf load_frag_a_f32(const float* p, int row0,
                                                 int k0, int ld, int lane) {
    const int m = lane & 15;
    const int h = lane >> 4;
    const float* base = p + (size_t)(row0 + m) * ld + k0 + 8 * h;
    v4f a0 = *(const v4f*)(base + 0);
    v4f a1 = *(const v4f*)(base + 4);
    v4f a2 = *(const v4f*)(base + 16);
    v4f a3 = *(const v4f*)(base + 20);
    union { v16bf v; bf16_t e[16]; } f;
#pragma unroll
    for (int j = 0; j < 4; ++j) {
        f.e[j]      = (bf16_t)a0[j];
        f.e[4 + j]  = (bf16_t)a1[j];
        f.e[8 + j]  = (bf16_t)a2[j];
        f.e[12 + j] = (bf16_t)a3[j];
    }
    return f.v;
}

// B fragment 32x16 where B = X^T and X is row-major bf16 [N, K]:
// lane L: col n = col0+(L&15); K = k0 + 8*(L>>4) + 0..7 and k0 + 16 + 8*(L>>4) + 0..7.
__device__ __forceinline__ v16bf load_frag_b_t(const bf16_t* p, int col0,
                                               int k0, int ld, int lane) {
    const int n = lane & 15;
    const int h = lane >> 4;
    const bf16_t* base = p + (size_t)(col0 + n) * ld + k0 + 8 * h;
    union { v16bf v; uint4 u[2]; } f;
    f.u[0] = *(const uint4*)(base);
    f.u[1] = *(const uint4*)(base + 16);
    return f.v;
}

// ---------------------------------------------------------------------------
// Kernel 0: WqT[n][k] = (bf16)Wq[k][n]  (1024x1024), LDS tile transpose.
// block (32,8), tile 32x32, grid (32,32).
// ---------------------------------------------------------------------------
__global__ __launch_bounds__(256) void wq_transpose(const float* __restrict__ Wq,
                                                    bf16_t* __restrict__ WqT) {
    __shared__ bf16_t tile[32][33];
    const int x0 = blockIdx.x * 32;   // n
    const int y0 = blockIdx.y * 32;   // k
    const int tx = threadIdx.x;
#pragma unroll
    for (int yy = threadIdx.y; yy < 32; yy += 8)
        tile[yy][tx] = (bf16_t)Wq[(size_t)(y0 + yy) * ATT_D + x0 + tx];
    __syncthreads();
#pragma unroll
    for (int yy = threadIdx.y; yy < 32; yy += 8)
        WqT[(size_t)(x0 + yy) * ATT_D + y0 + tx] = tile[tx][yy];
}

// ---------------------------------------------------------------------------
// Kernel 1: q = x @ Wq + bq; writes q_bf [16384,1024] and qT_bf [B][1024][2048].
// block (32,8); wave tile = 64(M) x 16(N); block tile = 512(M) x 16(N).
// grid (16384/512=32, 1024/16=64).
// ---------------------------------------------------------------------------
__global__ __launch_bounds__(256) void proj_kernel(const float* __restrict__ x,
                                                   const bf16_t* __restrict__ WqT,
                                                   const float* __restrict__ bq,
                                                   bf16_t* __restrict__ qout,
                                                   bf16_t* __restrict__ qT) {
    const int lane = threadIdx.x;
    const int wave = threadIdx.y;
    const int m0 = blockIdx.x * 512 + wave * 64;   // global row (b*S + s)
    const int n0 = blockIdx.y * 16;                // d column

    v8f c[4] = {};
#pragma unroll 4
    for (int k0 = 0; k0 < ATT_D; k0 += 32) {
        v16bf bf = load_frag_b_t(WqT, n0, k0, ATT_D, lane);
#pragma unroll
        for (int i = 0; i < 4; ++i) {
            v16bf a = load_frag_a_f32(x, m0 + 16 * i, k0, ATT_D, lane);
            c[i] = wmma_bf16(a, bf, c[i]);
        }
    }
    const int nl = lane & 15;
    const int h = lane >> 4;
    const int n = n0 + nl;
    const float bias = bq[n];
    const int b  = m0 >> 11;          // batch (block tile fits in one batch)
    const int s0 = (m0 & (ATT_S - 1)) + 8 * h;

#pragma unroll
    for (int i = 0; i < 4; ++i) {
        union { uint4 u; bf16_t e[8]; } pk;
#pragma unroll
        for (int v = 0; v < 8; ++v) {
            const float r = c[i][v] + bias;
            pk.e[v] = (bf16_t)r;
            qout[(size_t)(m0 + 16 * i + v + 8 * h) * ATT_D + n] = (bf16_t)r;
        }
        // qT[b][n][s]: the 8 rows per lane are contiguous in s -> one b128 store
        *(uint4*)(qT + ((size_t)b * ATT_D + n) * ATT_S + s0 + 16 * i) = pk.u;
    }
}

// ---------------------------------------------------------------------------
// Kernel 2: scores[b] = q_b @ q_b^T  (2048x2048, bf16 out).
// block (32,8); wave tile 64x64; block tile 128(M) x 256(N).
// grid (16, 8, 8).
// ---------------------------------------------------------------------------
__global__ __launch_bounds__(256) void scores_kernel(const bf16_t* __restrict__ q,
                                                     bf16_t* __restrict__ sc) {
    const int lane = threadIdx.x;
    const int wave = threadIdx.y;
    const int b = blockIdx.z;
    const int wm = wave >> 2, wn = wave & 3;
    const int m0 = blockIdx.x * 128 + wm * 64;
    const int n0 = blockIdx.y * 256 + wn * 64;
    const bf16_t* qb = q + (size_t)b * ATT_S * ATT_D;
    bf16_t* scb = sc + (size_t)b * ATT_S * ATT_S;

    v8f c[4][4] = {};
#pragma unroll 2
    for (int k0 = 0; k0 < ATT_D; k0 += 32) {
        __builtin_prefetch(qb + (size_t)(m0 + (lane & 15)) * ATT_D + k0 + 64, 0, 1);
        __builtin_prefetch(qb + (size_t)(n0 + (lane & 15)) * ATT_D + k0 + 64, 0, 1);
        v16bf a[4], bb[4];
#pragma unroll
        for (int i = 0; i < 4; ++i) a[i] = load_frag_a_bf16(qb, m0 + 16 * i, k0, ATT_D, lane);
#pragma unroll
        for (int j = 0; j < 4; ++j) bb[j] = load_frag_b_t(qb, n0 + 16 * j, k0, ATT_D, lane);
#pragma unroll
        for (int i = 0; i < 4; ++i)
#pragma unroll
            for (int j = 0; j < 4; ++j)
                c[i][j] = wmma_bf16(a[i], bb[j], c[i][j]);
    }
    const int nl = lane & 15;
    const int h = lane >> 4;
#pragma unroll
    for (int i = 0; i < 4; ++i)
#pragma unroll
        for (int j = 0; j < 4; ++j)
#pragma unroll
            for (int v = 0; v < 8; ++v)
                scb[(size_t)(m0 + 16 * i + v + 8 * h) * ATT_S + n0 + 16 * j + nl] =
                    (bf16_t)c[i][j][v];
}

// ---------------------------------------------------------------------------
// Kernel 3: row softmax in place over bf16 scores (one block per row).
// ---------------------------------------------------------------------------
__global__ __launch_bounds__(256) void softmax_rows(bf16_t* __restrict__ sc) {
    const size_t row = blockIdx.x;
    bf16_t* r = sc + row * ATT_S;
    __shared__ float red[256];
    const int t = threadIdx.x;

    float mx = -3.0e38f;
    for (int i = t; i < ATT_S; i += 256) mx = fmaxf(mx, (float)r[i]);
    red[t] = mx;
    __syncthreads();
    for (int s = 128; s > 0; s >>= 1) {
        if (t < s) red[t] = fmaxf(red[t], red[t + s]);
        __syncthreads();
    }
    mx = red[0];
    __syncthreads();

    float sum = 0.0f;
    for (int i = t; i < ATT_S; i += 256) sum += __expf((float)r[i] - mx);
    red[t] = sum;
    __syncthreads();
    for (int s = 128; s > 0; s >>= 1) {
        if (t < s) red[t] += red[t + s];
        __syncthreads();
    }
    const float inv = 1.0f / red[0];

    for (int i = t; i < ATT_S; i += 256)
        r[i] = (bf16_t)(__expf((float)r[i] - mx) * inv);
}

// ---------------------------------------------------------------------------
// Kernel 4: z[b] = attn_b (2048x2048 bf16) @ q_b -> f32, using qT for B.
// block (32,8); wave tile 64x64; block tile 128(M) x 256(N).
// grid (16, 4, 8).
// ---------------------------------------------------------------------------
__global__ __launch_bounds__(256) void pv_kernel(const bf16_t* __restrict__ sc,
                                                 const bf16_t* __restrict__ qT,
                                                 float* __restrict__ out) {
    const int lane = threadIdx.x;
    const int wave = threadIdx.y;
    const int b = blockIdx.z;
    const int wm = wave >> 2, wn = wave & 3;
    const int m0 = blockIdx.x * 128 + wm * 64;
    const int n0 = blockIdx.y * 256 + wn * 64;
    const bf16_t* scb = sc + (size_t)b * ATT_S * ATT_S;
    const bf16_t* qTb = qT + (size_t)b * ATT_D * ATT_S;   // [D][S]
    float* ob = out + (size_t)b * ATT_S * ATT_D;

    v8f c[4][4] = {};
#pragma unroll 2
    for (int k0 = 0; k0 < ATT_S; k0 += 32) {
        __builtin_prefetch(scb + (size_t)(m0 + (lane & 15)) * ATT_S + k0 + 64, 0, 1);
        __builtin_prefetch(qTb + (size_t)(n0 + (lane & 15)) * ATT_S + k0 + 64, 0, 1);
        v16bf a[4], bb[4];
#pragma unroll
        for (int i = 0; i < 4; ++i) a[i] = load_frag_a_bf16(scb, m0 + 16 * i, k0, ATT_S, lane);
#pragma unroll
        for (int j = 0; j < 4; ++j) bb[j] = load_frag_b_t(qTb, n0 + 16 * j, k0, ATT_S, lane);
#pragma unroll
        for (int i = 0; i < 4; ++i)
#pragma unroll
            for (int j = 0; j < 4; ++j)
                c[i][j] = wmma_bf16(a[i], bb[j], c[i][j]);
    }
    const int nl = lane & 15;
    const int h = lane >> 4;
#pragma unroll
    for (int i = 0; i < 4; ++i)
#pragma unroll
        for (int j = 0; j < 4; ++j)
#pragma unroll
            for (int v = 0; v < 8; ++v)
                ob[(size_t)(m0 + 16 * i + v + 8 * h) * ATT_D + n0 + 16 * j + nl] =
                    c[i][j][v];
}

// ---------------------------------------------------------------------------
extern "C" void kernel_launch(void* const* d_in, const int* in_sizes, int n_in,
                              void* d_out, int out_size, void* d_ws, size_t ws_size,
                              hipStream_t stream) {
    const float* x  = (const float*)d_in[0];   // [8,2048,1024]
    const float* Wq = (const float*)d_in[1];   // [1024,1024]
    const float* bq = (const float*)d_in[2];   // [1024]
    float* out = (float*)d_out;                // [8,2048,1024] f32

    // workspace layout: q_bf (32 MB) | qT_bf (32 MB) | WqT (2 MB) | scores (64 MB)
    char* ws = (char*)d_ws;
    bf16_t* q_bf  = (bf16_t*)(ws);
    bf16_t* qT_bf = (bf16_t*)(ws + (size_t)ATT_B * ATT_S * ATT_D * 2);
    bf16_t* WqT   = (bf16_t*)(ws + (size_t)ATT_B * ATT_S * ATT_D * 4);
    bf16_t* sc    = (bf16_t*)(ws + (size_t)ATT_B * ATT_S * ATT_D * 4 +
                              (size_t)ATT_D * ATT_D * 2);

    dim3 blk(32, 8);
    wq_transpose<<<dim3(32, 32), blk, 0, stream>>>(Wq, WqT);
    proj_kernel<<<dim3(32, 64), blk, 0, stream>>>(x, WqT, bq, q_bf, qT_bf);
    scores_kernel<<<dim3(16, 8, ATT_B), blk, 0, stream>>>(q_bf, sc);
    softmax_rows<<<dim3(ATT_B * ATT_S), dim3(256), 0, stream>>>(sc);
    pv_kernel<<<dim3(16, 4, ATT_B), blk, 0, stream>>>(sc, qT_bf, out);
}
